// F_att_real_1803886264330
// MI455X (gfx1250) — matrix operand order repaired and checked by reading, not compiled, not run on hardware
//
#include <hip/hip_runtime.h>

// Problem constants (match reference)
#define B_ 8
#define T_ 2048
#define C_ 512
#define H_ 512
#define M_ (B_ * T_)   // 16384 flattened rows

typedef __attribute__((ext_vector_type(16))) __bf16 bf16x16;
typedef __attribute__((ext_vector_type(8)))  float  f32x8;
typedef unsigned int u32x4 __attribute__((ext_vector_type(4)));
typedef int          i32x4 __attribute__((ext_vector_type(4)));
typedef int          i32x8 __attribute__((ext_vector_type(8)));

union FragA { uint4 u[2]; bf16x16 v; };

static __device__ inline f32x8 wmma_bf16(bf16x16 a, bf16x16 b, f32x8 c) {
  // D = A(16x32) * B(32x16) + C(16x16 f32)
  return __builtin_amdgcn_wmma_f32_16x16x32_bf16(
      /*neg_a=*/false, a, /*neg_b=*/false, b,
      /*c_mod=*/(short)0, c, /*reuse_a=*/false, /*reuse_b=*/false);
}

// ---------------------------------------------------------------------------
// Tensor Data Mover support (probe-verified builtin on ROCm 7.2: 5 args)
// ---------------------------------------------------------------------------
#if __has_builtin(__builtin_amdgcn_tensor_load_to_lds) && \
    __has_builtin(__builtin_amdgcn_s_wait_tensorcnt)
  #define USE_TDM 1
#else
  #define USE_TDM 0
#endif

#if USE_TDM
#if __has_include(<hip/amd_detail/amd_gfx1250_TDM.h>)
  #define TDM_6ARG 1   // therock-10.0 headers => 6-arg builtin
#else
  #define TDM_6ARG 0
#endif

// 2D bf16 tile load: tileW elems per row, tileH rows, row stride in elems.
// LDS padding: padI/padA per ISA 8.4 (interval code: 7=256 dwords, 3=16 dwords;
// amount code: 3 = 4 dwords = 16 bytes) -> reproduces banked LDS pitches.
static __device__ inline void tdm_load_2d(unsigned lds_byte_off, const void* gptr,
                                          unsigned tileW, unsigned tileH,
                                          unsigned long long strideElems,
                                          unsigned padI, unsigned padA)
{
  const unsigned long long ga = (unsigned long long)(__SIZE_TYPE__)gptr;
  u32x4 g0;
  g0.x = 1u;                                           // count=1, user desc
  g0.y = lds_byte_off;                                 // lds_addr
  g0.z = (unsigned)(ga & 0xFFFFFFFFull);               // global_addr[31:0]
  g0.w = (unsigned)((ga >> 32) & 0x1FFFFFFull) | (2u << 30);  // [56:32] | type=2
  i32x8 g1;
  g1[0] = (int)((1u << 16) |                           // data_size = 2 bytes
                (1u << 20) |                           // pad_enable
                (padI << 22) | (padA << 25));
  g1[1] = (int)((tileW & 0xFFFFu) << 16);              // tensor_dim0[15:0]
  g1[2] = (int)((tileW >> 16) | ((tileH & 0xFFFFu) << 16)); // dim0 hi | dim1 lo
  g1[3] = (int)((tileH >> 16) | (tileW << 16));        // dim1 hi | tile_dim0
  g1[4] = (int)(tileH & 0xFFFFu);                      // tile_dim1 (tile_dim2=0)
  g1[5] = (int)(strideElems & 0xFFFFFFFFull);          // tensor_dim0_stride lo
  g1[6] = (int)((strideElems >> 32) & 0xFFFFull);      // stride hi | dim1_stride=0
  g1[7] = 0;
  i32x4 g2 = {1, 1, 0, 0};                 // tensor_dim2=1, tensor_dim3=1
  i32x4 g3 = {0, (int)(1u << 16), 0, 0};   // tensor_dim4=1
#if TDM_6ARG
  i32x8 gz = {};
  __builtin_amdgcn_tensor_load_to_lds(g0, g1, g2, g3, gz, 0);
#else
  __builtin_amdgcn_tensor_load_to_lds(g0, g1, g2, g3, 0);
#endif
}
#endif  // USE_TDM

// ---------------------------------------------------------------------------
// Kernel 1: Y = (X @ W + bias) * scale, X:[M,C] f32, W:[C,H] f32 -> bf16
// transpose_out == 0 : Y row-major [M,H]
// transpose_out == 1 : Y = per-batch transpose [B][H][T]  (for V)
// ---------------------------------------------------------------------------
#define KP 48  // padded LDS pitch (bf16 elems): 96B rows, 16B-aligned frags

__global__ __launch_bounds__(256) void proj_gemm_kernel(
    const float* __restrict__ X, const float* __restrict__ W,
    const float* __restrict__ bias, __bf16* __restrict__ Y,
    float scale, int transpose_out)
{
  __shared__ __bf16 Xs[128 * KP];
  __shared__ __bf16 Ws[64 * KP];

  const int tid  = threadIdx.x;
  const int lane = tid & 31;
  const int wid  = tid >> 5;
  const int lm   = lane & 15;
  const int ph   = lane >> 4;
  const int wm   = wid >> 1;
  const int wn   = wid & 1;
  const int mBase = blockIdx.x * 128;
  const int nBase = blockIdx.y * 64;

  f32x8 acc[2][2] = {};

  for (int kk0 = 0; kk0 < C_; kk0 += 32) {
    __syncthreads();
    #pragma unroll
    for (int r4 = 0; r4 < 4; ++r4) {
      const int row = r4 * 32 + (tid >> 3);
      const int c4  = (tid & 7) * 4;
      const float4 x = *(const float4*)(X + (size_t)(mBase + row) * C_ + kk0 + c4);
      __bf16* d = &Xs[row * KP + c4];
      d[0] = (__bf16)x.x; d[1] = (__bf16)x.y; d[2] = (__bf16)x.z; d[3] = (__bf16)x.w;
    }
    #pragma unroll
    for (int r2 = 0; r2 < 2; ++r2) {
      const int k  = r2 * 16 + (tid >> 4);
      const int n4 = (tid & 15) * 4;
      const float4 w = *(const float4*)(W + (size_t)(kk0 + k) * H_ + nBase + n4);
      Ws[(n4 + 0) * KP + k] = (__bf16)w.x;
      Ws[(n4 + 1) * KP + k] = (__bf16)w.y;
      Ws[(n4 + 2) * KP + k] = (__bf16)w.z;
      Ws[(n4 + 3) * KP + k] = (__bf16)w.w;
    }
    __syncthreads();

    FragA a[2], b[2];
    #pragma unroll
    for (int sm = 0; sm < 2; ++sm) {
      const __bf16* src = &Xs[(wm * 32 + sm * 16 + lm) * KP];
      a[sm].u[0] = *(const uint4*)(src + ph * 8);
      a[sm].u[1] = *(const uint4*)(src + 16 + ph * 8);
    }
    #pragma unroll
    for (int sn = 0; sn < 2; ++sn) {
      const __bf16* src = &Ws[(wn * 32 + sn * 16 + lm) * KP];
      b[sn].u[0] = *(const uint4*)(src + ph * 8);
      b[sn].u[1] = *(const uint4*)(src + 16 + ph * 8);
    }
    #pragma unroll
    for (int sm = 0; sm < 2; ++sm)
      #pragma unroll
      for (int sn = 0; sn < 2; ++sn)
        acc[sm][sn] = wmma_bf16(a[sm].v, b[sn].v, acc[sm][sn]);
  }

  #pragma unroll
  for (int sm = 0; sm < 2; ++sm) {
    #pragma unroll
    for (int sn = 0; sn < 2; ++sn) {
      const int gcol = nBase + wn * 32 + sn * 16 + lm;
      const float bv = bias[gcol];
      const int row0 = mBase + wm * 32 + sm * 16 + ph * 8;
      if (!transpose_out) {
        #pragma unroll
        for (int i = 0; i < 8; ++i) {
          const float vv = (acc[sm][sn][i] + bv) * scale;
          Y[(size_t)(row0 + i) * H_ + gcol] = (__bf16)vv;
        }
      } else {
        unsigned pk[4];
        #pragma unroll
        for (int p = 0; p < 4; ++p) {
          const float v0 = (acc[sm][sn][2 * p]     + bv) * scale;
          const float v1 = (acc[sm][sn][2 * p + 1] + bv) * scale;
          const __bf16 b0 = (__bf16)v0, b1 = (__bf16)v1;
          const unsigned short u0 = __builtin_bit_cast(unsigned short, b0);
          const unsigned short u1 = __builtin_bit_cast(unsigned short, b1);
          pk[p] = (unsigned)u0 | ((unsigned)u1 << 16);
        }
        const int bidx = row0 >> 11;
        const int t0   = row0 & (T_ - 1);
        __bf16* dst = Y + ((size_t)(bidx * H_ + gcol)) * T_ + t0;
        *(uint4*)dst = make_uint4(pk[0], pk[1], pk[2], pk[3]);
      }
    }
  }
}

// ---------------------------------------------------------------------------
// Kernel 2: flash attention with TDM double-buffered K/V staging.
// Block: 256 thr (8 waves) -> 64 q-rows of one batch.
// wave = (wid&3) picks 16 q-rows; (wid>>2) picks H-half (256 cols).
// KV chunk = 32 keys.
// ---------------------------------------------------------------------------
#define QPITCH 520   // 512 + 8 pad (bf16 elems)
#define VPITCH 40    // 32 + 8 pad
#define PPITCH 48    // 32 + 16 pad
#define NCH    (T_ / 32)

#define QS_ELEMS (64 * QPITCH)
#define KS_ELEMS (32 * QPITCH)
#define VS_ELEMS (512 * VPITCH)
#define PS_ELEMS (8 * 16 * PPITCH)
// layout: Qs | Ks[0] | Ks[1] | Vs[0] | Vs[1] | Ps   (double-buffered K/V)
#define SMEM_BYTES ((QS_ELEMS + 2 * KS_ELEMS + 2 * VS_ELEMS + PS_ELEMS) * 2)

__global__ __launch_bounds__(256, 1) void attn_flash_kernel(
    const __bf16* __restrict__ Qb, const __bf16* __restrict__ Kb,
    const __bf16* __restrict__ Vt, float* __restrict__ Out)
{
  extern __shared__ __bf16 smem[];
  __bf16* Qs = smem;
  __bf16* Ks = Qs + QS_ELEMS;             // 2 buffers
  __bf16* Vs = Ks + 2 * KS_ELEMS;         // 2 buffers
  __bf16* Ps = Vs + 2 * VS_ELEMS;

  const int tid  = threadIdx.x;
  const int lane = tid & 31;
  const int wid  = tid >> 5;
  const int lm   = lane & 15;
  const int ph   = lane >> 4;
  const int wq   = wid & 3;
  const int wh   = wid >> 2;
  const int b    = blockIdx.y;
  const int q0   = blockIdx.x * 64;
  const size_t qrowg = (size_t)b * T_ + q0;

#if USE_TDM
  // LDS byte offsets of the dynamic-shared partitions (dyn base = static size)
  const unsigned ldsBase = (unsigned)__builtin_amdgcn_groupstaticsize();
  const unsigned offQ  = ldsBase;
  const unsigned offK0 = ldsBase + QS_ELEMS * 2;
  const unsigned offV0 = ldsBase + (QS_ELEMS + 2 * KS_ELEMS) * 2;

  if (wid == 0) {
    // Q block: 64 rows x 512 elems, contiguous rows; pad 16B per 1024B row
    tdm_load_2d(offQ, Qb + qrowg * H_, 512, 64, 512, /*padI=*/7, /*padA=*/3);
    // chunk 0: K 32x512 rows; V^T 512x32 (row stride T_)
    tdm_load_2d(offK0, Kb + (size_t)b * T_ * H_, 512, 32, 512, 7, 3);
    tdm_load_2d(offV0, Vt + (size_t)b * H_ * T_, 32, 512, T_, 3, 3);
    __builtin_amdgcn_s_wait_tensorcnt(0);
  }
  __syncthreads();
#else
  // cooperative copies (fallback)
  #pragma unroll
  for (int it = 0; it < 16; ++it) {
    const int u   = tid + it * 256;
    const int row = u >> 6;
    const int c   = (u & 63) * 8;
    const uint4 x = *(const uint4*)(Qb + (qrowg + row) * H_ + c);
    *(uint4*)(Qs + row * QPITCH + c) = x;
  }
  #pragma unroll
  for (int it = 0; it < 8; ++it) {
    const int u   = tid + it * 256;
    const int row = u >> 6;
    const int c   = (u & 63) * 8;
    const uint4 x = *(const uint4*)(Kb + ((size_t)b * T_ + row) * H_ + c);
    *(uint4*)(Ks + row * QPITCH + c) = x;
  }
  #pragma unroll
  for (int it = 0; it < 8; ++it) {
    const int u = tid + it * 256;
    const int h = u >> 2;
    const int c = (u & 3) * 8;
    const uint4 x = *(const uint4*)(Vt + ((size_t)b * H_ + h) * T_ + c);
    *(uint4*)(Vs + h * VPITCH + c) = x;
  }
  __syncthreads();
#endif

  float mrow[8], lrow[8];
  #pragma unroll
  for (int i = 0; i < 8; ++i) { mrow[i] = -3.0e38f; lrow[i] = 0.0f; }
  f32x8 o[16] = {};

  __bf16* Pw = Ps + wid * (16 * PPITCH);

  for (int c = 0; c < NCH; ++c) {
    const int buf = c & 1;
    const __bf16* KsC = Ks + buf * KS_ELEMS;
    const __bf16* VsC = Vs + buf * VS_ELEMS;

    // ---- prefetch chunk c+1 into the other buffer ----
    if (c + 1 < NCH) {
      const int kvn = (c + 1) * 32;
      const int nb  = buf ^ 1;
#if USE_TDM
      if (wid == 0) {
        tdm_load_2d(offK0 + nb * (KS_ELEMS * 2),
                    Kb + ((size_t)b * T_ + kvn) * H_, 512, 32, 512, 7, 3);
        tdm_load_2d(offV0 + nb * (VS_ELEMS * 2),
                    Vt + (size_t)b * H_ * T_ + kvn, 32, 512, T_, 3, 3);
      }
#else
      __bf16* Kn = Ks + nb * KS_ELEMS;
      __bf16* Vn = Vs + nb * VS_ELEMS;
      #pragma unroll
      for (int it = 0; it < 8; ++it) {
        const int u   = tid + it * 256;
        const int row = u >> 6;
        const int cc  = (u & 63) * 8;
        const uint4 x = *(const uint4*)(Kb + ((size_t)b * T_ + kvn + row) * H_ + cc);
        *(uint4*)(Kn + row * QPITCH + cc) = x;
      }
      #pragma unroll
      for (int it = 0; it < 8; ++it) {
        const int u  = tid + it * 256;
        const int h  = u >> 2;
        const int cc = (u & 3) * 8;
        const uint4 x = *(const uint4*)(Vt + ((size_t)b * H_ + h) * T_ + kvn + cc);
        *(uint4*)(Vn + h * VPITCH + cc) = x;
      }
#endif
    }

    // ---- S = Q Kt : 16 rows x 32 keys, accumulate over H=512 ----
    f32x8 s[2] = {};
    #pragma unroll
    for (int hk = 0; hk < 16; ++hk) {
      FragA a;
      const __bf16* qs = Qs + (wq * 16 + lm) * QPITCH + hk * 32;
      a.u[0] = *(const uint4*)(qs + ph * 8);
      a.u[1] = *(const uint4*)(qs + 16 + ph * 8);
      #pragma unroll
      for (int sn = 0; sn < 2; ++sn) {
        FragA bb;
        const __bf16* ks = KsC + (sn * 16 + lm) * QPITCH + hk * 32;
        bb.u[0] = *(const uint4*)(ks + ph * 8);
        bb.u[1] = *(const uint4*)(ks + 16 + ph * 8);
        s[sn] = wmma_bf16(a.v, bb.v, s[sn]);
      }
    }

    // ---- online softmax ----
    float mnew[8], alpha[8], p0[8], p1[8];
    #pragma unroll
    for (int i = 0; i < 8; ++i) {
      float ml = fmaxf(s[0][i], s[1][i]);
      #pragma unroll
      for (int off = 8; off >= 1; off >>= 1)
        ml = fmaxf(ml, __shfl_xor(ml, off, 32));
      mnew[i]  = fmaxf(mrow[i], ml);
      alpha[i] = __expf(mrow[i] - mnew[i]);
      mrow[i]  = mnew[i];
    }
    #pragma unroll
    for (int i = 0; i < 8; ++i) {
      p0[i] = __expf(s[0][i] - mnew[i]);
      p1[i] = __expf(s[1][i] - mnew[i]);
      float rs = p0[i] + p1[i];
      #pragma unroll
      for (int off = 8; off >= 1; off >>= 1)
        rs += __shfl_xor(rs, off, 32);
      lrow[i] = lrow[i] * alpha[i] + rs;
    }

    // ---- P (D-layout) -> LDS -> A-layout fragment ----
    #pragma unroll
    for (int i = 0; i < 8; ++i) {
      const int r = ph * 8 + i;
      Pw[r * PPITCH + lm]      = (__bf16)p0[i];
      Pw[r * PPITCH + 16 + lm] = (__bf16)p1[i];
    }
#if defined(__HIP_DEVICE_COMPILE__)
    asm volatile("s_wait_dscnt 0" ::: "memory");
#endif
    FragA pf;
    {
      const __bf16* ps = Pw + lm * PPITCH;
      pf.u[0] = *(const uint4*)(ps + ph * 8);
      pf.u[1] = *(const uint4*)(ps + 16 + ph * 8);
    }

    // ---- O = O*alpha + P @ V (this wave's 256-col H half) ----
    #pragma unroll
    for (int nt = 0; nt < 16; ++nt) {
      #pragma unroll
      for (int i = 0; i < 8; ++i) o[nt][i] *= alpha[i];
      FragA vb;
      const __bf16* vs = VsC + (wh * 256 + nt * 16 + lm) * VPITCH;
      vb.u[0] = *(const uint4*)(vs + ph * 8);
      vb.u[1] = *(const uint4*)(vs + 16 + ph * 8);
      o[nt] = wmma_bf16(pf.v, vb.v, o[nt]);
    }

    // ---- close pipeline stage: prefetch done + all waves done with buf ----
#if USE_TDM
    if (wid == 0) __builtin_amdgcn_s_wait_tensorcnt(0);
#endif
    __syncthreads();
  }

  // ---- normalize and store f32 output [B,T,H] ----
  float inv[8];
  #pragma unroll
  for (int i = 0; i < 8; ++i) inv[i] = 1.0f / lrow[i];
  const size_t orow0 = (size_t)b * T_ + q0 + wq * 16 + ph * 8;
  #pragma unroll
  for (int nt = 0; nt < 16; ++nt) {
    const int col = wh * 256 + nt * 16 + lm;
    #pragma unroll
    for (int i = 0; i < 8; ++i)
      Out[(orow0 + i) * H_ + col] = o[nt][i] * inv[i];
  }
}

// ---------------------------------------------------------------------------
extern "C" void kernel_launch(void* const* d_in, const int* in_sizes, int n_in,
                              void* d_out, int out_size, void* d_ws, size_t ws_size,
                              hipStream_t stream) {
  const float* q  = (const float*)d_in[0];
  const float* k  = (const float*)d_in[1];
  const float* v  = (const float*)d_in[2];
  const float* Wq = (const float*)d_in[3];
  const float* bq = (const float*)d_in[4];
  const float* Wk = (const float*)d_in[5];
  const float* bk = (const float*)d_in[6];
  const float* Wv = (const float*)d_in[7];
  const float* bv = (const float*)d_in[8];
  float* out = (float*)d_out;

  // workspace: Qb [M,H] bf16 | Kb [M,H] bf16 | Vt [B,H,T] bf16  (48 MB total)
  __bf16* Qb = (__bf16*)d_ws;
  __bf16* Kb = Qb + (size_t)M_ * H_;
  __bf16* Vt = Kb + (size_t)M_ * H_;

  const float qscale = 0.04419417382415922f;  // 1/sqrt(512) folded into Q

  dim3 gemm_grid(M_ / 128, H_ / 64);
  proj_gemm_kernel<<<gemm_grid, 256, 0, stream>>>(q, Wq, bq, Qb, qscale, 0);
  proj_gemm_kernel<<<gemm_grid, 256, 0, stream>>>(k, Wk, bk, Kb, 1.0f,  0);
  proj_gemm_kernel<<<gemm_grid, 256, 0, stream>>>(v, Wv, bv, Vt, 1.0f,  1);

  (void)hipFuncSetAttribute((const void*)attn_flash_kernel,
                            hipFuncAttributeMaxDynamicSharedMemorySize,
                            SMEM_BYTES);
  dim3 attn_grid(T_ / 64, B_);
  attn_flash_kernel<<<attn_grid, 256, SMEM_BYTES, stream>>>(Qb, Kb, Vt, out);
}
